// SpatiotemporalAttentionBase_16114717295099
// MI455X (gfx1250) — matrix operand order repaired and checked by reading, not compiled, not run on hardware
//
#include <hip/hip_runtime.h>
#include <math.h>

// Problem constants (match reference)
#define N_POS 4096   // H*W
#define C_IN  64     // C
#define C_I   32     // CI
#define BATCH 4
#define BN_EPS 1e-5f

typedef __attribute__((ext_vector_type(16))) _Float16 v16h;
typedef __attribute__((ext_vector_type(8)))  _Float16 v8h;
typedef __attribute__((ext_vector_type(2)))  __fp16   v2fp16;
typedef __attribute__((ext_vector_type(8)))  float    v8f;
typedef __attribute__((ext_vector_type(8)))  unsigned v8u;

__device__ __forceinline__ v8f zero8() {
  v8f z;
#pragma unroll
  for (int i = 0; i < 8; ++i) z[i] = 0.f;
  return z;
}

__device__ __forceinline__ v16h pack16(v8h lo, v8h hi) {
  v16h a;
#pragma unroll
  for (int h = 0; h < 8; ++h) { a[h] = lo[h]; a[h + 8] = hi[h]; }
  return a;
}

__device__ __forceinline__ unsigned packh2(float a, float b) {
  v2fp16 v = __builtin_amdgcn_cvt_pkrtz(a, b);   // low half = a
  return __builtin_bit_cast(unsigned, v);
}

__device__ __forceinline__ v8f wmma_f16(v16h a, v16h b, v8f c) {
  // D = A(16x32 f16) * B(32x16 f16) + C(16x16 f32)
  return __builtin_amdgcn_wmma_f32_16x16x32_f16(
      /*neg_a=*/false, a, /*neg_b=*/false, b,
      /*c_mod=*/(short)0, c, /*reuse_a=*/false, /*reuse_b=*/false);
}

// ---------------------------------------------------------------------------
// Kernel 1: fold eval-mode BN into the 1x1 conv weights for g/theta/phi.
// ---------------------------------------------------------------------------
__global__ void prep_kernel(const float* __restrict__ gbn, const float* __restrict__ gw,
                            const float* __restrict__ gb,
                            const float* __restrict__ tbn, const float* __restrict__ tw,
                            const float* __restrict__ tb,
                            const float* __restrict__ pbn, const float* __restrict__ pw,
                            const float* __restrict__ pb,
                            float* __restrict__ Wf, float* __restrict__ Bf) {
  int t = threadIdx.x;
  if (t >= 3 * C_I) return;
  int p = t >> 5, ci = t & (C_I - 1);
  const float* bn = (p == 0) ? gbn : ((p == 1) ? tbn : pbn);
  const float* w  = (p == 0) ? gw  : ((p == 1) ? tw  : pw);
  const float* bb = (p == 0) ? gb  : ((p == 1) ? tb  : pb);
  float acc = bb[ci];
  for (int c = 0; c < C_IN; ++c) {
    float gamma = bn[c], beta = bn[64 + c], mean = bn[128 + c], var = bn[192 + c];
    float inv = gamma / sqrtf(var + BN_EPS);
    float wv = w[ci * C_IN + c];
    Wf[p * C_I * C_IN + ci * C_IN + c] = wv * inv;
    acc += wv * (beta - mean * inv);
  }
  Bf[p * C_I + ci] = acc;
}

// ---------------------------------------------------------------------------
// Kernel 2: projections (BN-folded 1x1 conv) -> f16 [CI][N]; theta/phi also
// write a transposed copy [N][CI] so attend's A-tiles are aligned b128 loads.
// blockIdx.z: 0 = g(x1)->g1, 1 = g(x2)->g2, 2 = theta(x1)->th, 3 = phi(x2)->ph
// ---------------------------------------------------------------------------
__global__ void project_kernel(const float* __restrict__ x1, const float* __restrict__ x2,
                               const float* __restrict__ Wf, const float* __restrict__ Bf,
                               _Float16* __restrict__ g1, _Float16* __restrict__ g2,
                               _Float16* __restrict__ th, _Float16* __restrict__ ph,
                               _Float16* __restrict__ thT, _Float16* __restrict__ phT) {
  __shared__ float sx[C_IN * 128];
  __shared__ float sw[C_I * C_IN];
  __shared__ float sb[C_I];
  int t = threadIdx.x;                       // 0..127
  int n0 = blockIdx.x * 128;
  int b = blockIdx.y;
  int z = blockIdx.z;
  const float* x = (z == 0 || z == 2) ? x1 : x2;
  int wsel = (z < 2) ? 0 : (z - 1);          // g, theta, phi
  const float* xb = x + (size_t)b * C_IN * N_POS;
  for (int c = 0; c < C_IN; ++c)
    sx[c * 128 + t] = xb[(size_t)c * N_POS + n0 + t];
  for (int idx = t; idx < C_I * C_IN; idx += 128)
    sw[idx] = Wf[wsel * C_I * C_IN + idx];
  if (t < C_I) sb[t] = Bf[wsel * C_I + t];
  __syncthreads();

  _Float16* out  = (z == 0) ? g1 : ((z == 1) ? g2 : ((z == 2) ? th : ph));
  _Float16* outT = (z == 2) ? thT : ((z == 3) ? phT : (_Float16*)0);
  _Float16* ob = out + (size_t)b * C_I * N_POS;
  _Float16* obT = outT ? outT + (size_t)b * N_POS * C_I : (_Float16*)0;

  for (int ci = 0; ci < C_I; ++ci) {
    float acc = sb[ci];
#pragma unroll
    for (int c = 0; c < C_IN; ++c)
      acc += sw[ci * C_IN + c] * sx[c * 128 + t];
    _Float16 hv = (_Float16)acc;
    ob[(size_t)ci * N_POS + n0 + t] = hv;
    if (obT) obT[(size_t)(n0 + t) * C_I + ci] = hv;
  }
}

// ---------------------------------------------------------------------------
// Kernel 3: fused "flash" attention sweep (no LDS; partner-lane exchange).
//   E'[i,j] = sum_k U[k][i] V[k][j]   (K = CI = 32 -> one WMMA per 16x16)
//   att     = online softmax over i, per-lane (column lives in a lane pair)
//   Y[c,j] += G(:,i-chunk) * P       (P relayout D->B = 4 ds_bpermute swaps)
// Each wave owns TWO 16-wide j-tiles (32 cols) so the j-independent A-tiles
// (UT rows, G rows) are amortized over 8 WMMAs per iteration.
// ---------------------------------------------------------------------------
__global__ void attend_kernel(const _Float16* __restrict__ UT,  // [B][N][CI]
                              const _Float16* __restrict__ V,   // [B][CI][N]
                              const _Float16* __restrict__ G,   // [B][CI][N]
                              float* __restrict__ Y) {          // [B][CI][N]
  const int lane = threadIdx.x & 31;
  const int wave = threadIdx.x >> 5;
  const int b = blockIdx.y;
  const int j0 = (blockIdx.x * 4 + wave) * 32;
  const int jn = lane & 15;                  // column index within a tile
  const bool lo16 = (lane < 16);
  const int abase = lo16 ? 0 : 16;           // A-layout K base (16-bit A 16x32)
  const int kb = lo16 ? 0 : 16;              // B-layout K base
  const int rowoff = lo16 ? 0 : 8;           // D-layout M base

  const _Float16* UTb = UT + (size_t)b * N_POS * C_I;
  const _Float16* Vb  = V + (size_t)b * C_I * N_POS;
  const _Float16* Gb  = G + (size_t)b * C_I * N_POS;
  float* Yb = Y + (size_t)b * C_I * N_POS;

  // Loop-invariant B tiles of V (two j-tiles)
  v16h bv[2];
#pragma unroll
  for (int t = 0; t < 2; ++t)
#pragma unroll
    for (int h = 0; h < 16; ++h)
      bv[t][h] = Vb[(size_t)(kb + h) * N_POS + j0 + t * 16 + jn];

  float m[2] = {-1e30f, -1e30f};   // running column max
  float s[2] = {0.f, 0.f};         // per-lane partial exp-sums
  v8f yA[2], yB[2];
  yA[0] = zero8(); yA[1] = zero8(); yB[0] = zero8(); yB[1] = zero8();

  for (int i0 = 0; i0 < N_POS; i0 += 32) {
    // A-tiles (j-independent): U^T rows i0..i0+31, G rows c=0..31
    const _Float16* r0 = UTb + (size_t)(i0 + jn) * C_I + abase;
    const _Float16* r1 = UTb + (size_t)(i0 + 16 + jn) * C_I + abase;
    v16h a0 = pack16(*(const v8h*)r0, *(const v8h*)(r0 + 8));
    v16h a1 = pack16(*(const v8h*)r1, *(const v8h*)(r1 + 8));
    const _Float16* gr0 = Gb + (size_t)jn * N_POS + i0 + abase;
    const _Float16* gr1 = Gb + (size_t)(16 + jn) * N_POS + i0 + abase;
    v16h ga0 = pack16(*(const v8h*)gr0, *(const v8h*)(gr0 + 8));
    v16h ga1 = pack16(*(const v8h*)gr1, *(const v8h*)(gr1 + 8));
    if (i0 + 32 < N_POS) {                 // prefetch next iteration's A rows
      __builtin_prefetch(r0 + 32 * C_I, 0, 1);
      __builtin_prefetch(gr0 + 32, 0, 1);
    }

#pragma unroll
    for (int t = 0; t < 2; ++t) {
      v8f e0 = wmma_f16(a0, bv[t], zero8());   // rows i0..i0+15
      v8f e1 = wmma_f16(a1, bv[t], zero8());   // rows i0+16..i0+31

      // online softmax update; column j lives in lanes {jn, jn+16}
      float lm = -1e30f;
#pragma unroll
      for (int r = 0; r < 8; ++r) { lm = fmaxf(lm, e0[r]); lm = fmaxf(lm, e1[r]); }
      lm = fmaxf(lm, __shfl_xor(lm, 16, 32));
      float mnew = fmaxf(m[t], lm);
      float scale = __expf(m[t] - mnew);       // first iter: exp(-huge) = 0
      m[t] = mnew;
      s[t] *= scale;
#pragma unroll
      for (int r = 0; r < 8; ++r) { yA[t][r] *= scale; yB[t][r] *= scale; }

      // P = exp(E - m); D->B relayout = swap one 8-half block between
      // partner lanes (4 packed-half2 bpermutes), no LDS needed.
      float p0[8], p1[8];
      float acc = 0.f;
#pragma unroll
      for (int r = 0; r < 8; ++r) {
        p0[r] = __expf(e0[r] - mnew); acc += p0[r];
        p1[r] = __expf(e1[r] - mnew); acc += p1[r];
      }
      s[t] += acc;
      unsigned opk[4], ppk[4], qpk[4];
#pragma unroll
      for (int d = 0; d < 4; ++d) {
        opk[d] = packh2(p0[2 * d], p0[2 * d + 1]);  // lane<16: k0..7 | lane>=16: k8..15
        ppk[d] = packh2(p1[2 * d], p1[2 * d + 1]);  // lane<16: k16..23 | lane>=16: k24..31
      }
#pragma unroll
      for (int d = 0; d < 4; ++d)
        qpk[d] = __shfl_xor(lo16 ? ppk[d] : opk[d], 16, 32);
      v8u bu;
#pragma unroll
      for (int d = 0; d < 4; ++d) {
        bu[d]     = lo16 ? opk[d] : qpk[d];   // k(kb+0..7)
        bu[d + 4] = lo16 ? qpk[d] : ppk[d];   // k(kb+8..15)
      }
      v16h bp = __builtin_bit_cast(v16h, bu);

      yA[t] = wmma_f16(ga0, bp, yA[t]);
      yB[t] = wmma_f16(ga1, bp, yB[t]);
    }
  }

  // finalize: full column sum = this lane's partial + partner's partial
#pragma unroll
  for (int t = 0; t < 2; ++t) {
    float S = s[t] + __shfl_xor(s[t], 16, 32);
    float invs = 1.f / S;
#pragma unroll
    for (int r = 0; r < 8; ++r) {
      Yb[(size_t)(r + rowoff) * N_POS + j0 + t * 16 + jn] = yA[t][r] * invs;
      Yb[(size_t)(16 + r + rowoff) * N_POS + j0 + t * 16 + jn] = yB[t][r] * invs;
    }
  }
}

// ---------------------------------------------------------------------------
// Kernel 4: out = x + BN_W( W_w @ y + W_b )   (per-frame epilogue, f32)
// ---------------------------------------------------------------------------
__global__ void epilogue_kernel(const float* __restrict__ x,
                                const float* __restrict__ Y,   // [B][CI][N]
                                const float* __restrict__ Ww,  // [C][CI]
                                const float* __restrict__ Wb,  // [C]
                                const float* __restrict__ wbn, // [4][C]
                                float* __restrict__ out) {
  int idx = blockIdx.x * blockDim.x + threadIdx.x;   // < B*C*N
  int p = idx & (N_POS - 1);
  int o = (idx >> 12) & (C_IN - 1);
  int b = idx >> 18;
  const float* Yb = Y + (size_t)b * C_I * N_POS;
  float acc = Wb[o];
#pragma unroll
  for (int c = 0; c < C_I; ++c)
    acc += Ww[o * C_I + c] * Yb[(size_t)c * N_POS + p];
  float gamma = wbn[o], beta = wbn[64 + o], mean = wbn[128 + o], var = wbn[192 + o];
  float inv = gamma / sqrtf(var + BN_EPS);
  out[idx] = x[idx] + acc * inv + (beta - mean * inv);
}

// ---------------------------------------------------------------------------
extern "C" void kernel_launch(void* const* d_in, const int* in_sizes, int n_in,
                              void* d_out, int out_size, void* d_ws, size_t ws_size,
                              hipStream_t stream) {
  (void)in_sizes; (void)n_in; (void)out_size; (void)ws_size;
  const float* x1  = (const float*)d_in[0];
  const float* x2  = (const float*)d_in[1];
  const float* gbn = (const float*)d_in[2];
  const float* gw  = (const float*)d_in[3];
  const float* gb  = (const float*)d_in[4];
  const float* tbn = (const float*)d_in[5];
  const float* tw  = (const float*)d_in[6];
  const float* tb  = (const float*)d_in[7];
  const float* pbn = (const float*)d_in[8];
  const float* pw  = (const float*)d_in[9];
  const float* pb  = (const float*)d_in[10];
  const float* wbn = (const float*)d_in[11];
  const float* Ww  = (const float*)d_in[12];
  const float* Wb  = (const float*)d_in[13];

  // Workspace layout (~10 MB total)
  char* ws = (char*)d_ws;
  const size_t MB = 1024 * 1024;
  _Float16* th  = (_Float16*)(ws + 0 * MB);   // [B][CI][N] f16
  _Float16* ph  = (_Float16*)(ws + 1 * MB);
  _Float16* g1  = (_Float16*)(ws + 2 * MB);
  _Float16* g2  = (_Float16*)(ws + 3 * MB);
  _Float16* thT = (_Float16*)(ws + 4 * MB);   // [B][N][CI] f16
  _Float16* phT = (_Float16*)(ws + 5 * MB);
  float* Y1 = (float*)(ws + 6 * MB);          // [B][CI][N] f32
  float* Y2 = (float*)(ws + 8 * MB);
  float* Wf = (float*)(ws + 10 * MB);         // folded conv weights
  float* Bf = (float*)(ws + 10 * MB + 32768);

  float* out1 = (float*)d_out;
  float* out2 = out1 + (size_t)BATCH * C_IN * N_POS;

  hipLaunchKernelGGL(prep_kernel, dim3(1), dim3(128), 0, stream,
                     gbn, gw, gb, tbn, tw, tb, pbn, pw, pb, Wf, Bf);
  hipLaunchKernelGGL(project_kernel, dim3(N_POS / 128, BATCH, 4), dim3(128), 0, stream,
                     x1, x2, Wf, Bf, g1, g2, th, ph, thT, phT);
  // y1: softmax over n of E[n,m];  y2: softmax over m of E[n,m] (via E^T)
  hipLaunchKernelGGL(attend_kernel, dim3(N_POS / 128, BATCH), dim3(128), 0, stream,
                     thT, ph, g1, Y1);
  hipLaunchKernelGGL(attend_kernel, dim3(N_POS / 128, BATCH), dim3(128), 0, stream,
                     phT, th, g2, Y2);
  hipLaunchKernelGGL(epilogue_kernel, dim3((BATCH * C_IN * N_POS) / 256), dim3(256), 0, stream,
                     x1, Y1, Ww, Wb, wbn, out1);
  hipLaunchKernelGGL(epilogue_kernel, dim3((BATCH * C_IN * N_POS) / 256), dim3(256), 0, stream,
                     x2, Y2, Ww, Wb, wbn, out2);
}
